// middle_attention_52536039965221
// MI455X (gfx1250) — matrix operand order repaired and checked by reading, not compiled, not run on hardware
//
#include <hip/hip_runtime.h>
#include <hip/hip_bf16.h>
#include <stdint.h>

typedef __attribute__((ext_vector_type(2)))  float  v2f;
typedef __attribute__((ext_vector_type(8)))  float  v8f;
typedef __attribute__((ext_vector_type(8)))  __bf16 v8bf;
typedef __attribute__((ext_vector_type(16))) __bf16 v16bf;

constexpr int Bn = 8, Sn = 2048, Hn = 128, NHn = 4, HSn = 32;
constexpr float RSCALE = 0.17677669529663688f; // 1/sqrt(HS)
constexpr float EPSL = 1e-8f;

__device__ __forceinline__ v8f wmma_f32(v2f a, v2f b, v8f c) {
  return __builtin_amdgcn_wmma_f32_16x16x4_f32(false, a, false, b, (short)0, c,
                                               false, false);
}
__device__ __forceinline__ v8f wmma_bf(v16bf a, v16bf b, v8f c) {
  return __builtin_amdgcn_wmma_f32_16x16x32_bf16(false, a, false, b, (short)0,
                                                 c, false, false);
}
__device__ __forceinline__ uint32_t lds_off(const void* p) {
  return (uint32_t)(uintptr_t)p;  // generic LDS pointer: low 32 bits = offset
}

// ---------------------------------------------------------------------------
// Kernel 1: projections Y = X @ W + b (exact f32 WMMA) -> split-bf16 planes.
//   Q,Qb (pre-scaled 1/sqrt(HS)), K,Kb : [B*NH][S][HS]
//   V                                  : [B*NH][HS][S]  (transposed for PV B)
// grid = (B*S/16, 5), block = 256 (8 waves)
// ---------------------------------------------------------------------------
__global__ __launch_bounds__(256) void proj_wmma_kernel(
    const float* __restrict__ item, const float* __restrict__ behav,
    const float* __restrict__ Wq, const float* __restrict__ bq,
    const float* __restrict__ Wk, const float* __restrict__ bk,
    const float* __restrict__ Wv, const float* __restrict__ bv,
    const float* __restrict__ Wqb, const float* __restrict__ bqb,
    const float* __restrict__ Wkb, const float* __restrict__ bkb,
    __bf16* __restrict__ Qhi, __bf16* __restrict__ Qlo,
    __bf16* __restrict__ Khi, __bf16* __restrict__ Klo,
    __bf16* __restrict__ Vthi, __bf16* __restrict__ Vtlo,
    __bf16* __restrict__ Qbhi, __bf16* __restrict__ Qblo,
    __bf16* __restrict__ Kbhi, __bf16* __restrict__ Kblo) {
  __shared__ float Xs[16][132];

  const int which = blockIdx.y;
  const float* X;
  const float* W;
  const float* bias;
  __bf16 *ohi, *olo;
  switch (which) {
    case 0: X = item;  W = Wq;  bias = bq;  ohi = Qhi;  olo = Qlo;  break;
    case 1: X = item;  W = Wk;  bias = bk;  ohi = Khi;  olo = Klo;  break;
    case 2: X = item;  W = Wv;  bias = bv;  ohi = Vthi; olo = Vtlo; break;
    case 3: X = behav; W = Wqb; bias = bqb; ohi = Qbhi; olo = Qblo; break;
    default:X = behav; W = Wkb; bias = bkb; ohi = Kbhi; olo = Kblo; break;
  }
  const bool isV = (which == 2);
  const bool doScale = (which == 0 || which == 3);

  const int rowbase = blockIdx.x * 16;
  for (int i = threadIdx.x; i < 16 * Hn; i += 256) {
    int r = i >> 7, c = i & 127;
    Xs[r][c] = X[(size_t)(rowbase + r) * Hn + c];
  }
  __syncthreads();

  const int lane = threadIdx.x & 31;
  const int lr = lane & 15;
  const int hi = lane >> 4;
  const int ncol = (threadIdx.x >> 5) * 16 + lr;

  v8f acc = {};
#pragma unroll
  for (int k = 0; k < Hn; k += 4) {
    v2f a, b;
    a.x = Xs[lr][k + 2 * hi];
    a.y = Xs[lr][k + 2 * hi + 1];
    b.x = W[(size_t)(k + 2 * hi) * Hn + ncol];
    b.y = W[(size_t)(k + 2 * hi + 1) * Hn + ncol];
    acc = wmma_f32(a, b, acc);
  }

  const float bv_ = bias[ncol];
  const int h = ncol >> 5, d = ncol & 31;
  const int b0 = rowbase / Sn;
#pragma unroll
  for (int v = 0; v < 8; v++) {
    int r = rowbase + v + 8 * hi;
    int s = r - b0 * Sn;
    float y = acc[v] + bv_;
    if (doScale) y *= RSCALE;
    __bf16 hb = (__bf16)y;
    __bf16 lb = (__bf16)(y - (float)hb);
    size_t idx = isV ? ((((size_t)b0 * NHn + h) * HSn + d) * Sn + s)
                     : ((((size_t)b0 * NHn + h) * Sn + s) * HSn + d);
    ohi[idx] = hb;
    olo[idx] = lb;
  }
}

// ---------------------------------------------------------------------------
// Kernel 2: flash attention, bf16x3 WMMA, async double-buffered LDS staging.
// Block = 4 waves sharing one (b,h); each wave owns a 16-query tile.
// Per 32-key iteration each wave issues 6 GLOBAL_LOAD_ASYNC_TO_LDS_B128
// copies for the NEXT buffer, waits s_wait_asynccnt 6 for the current one.
// grid = B*NH*(S/16)/4, block = 128. AO: [B][S][H] fp32.
// ---------------------------------------------------------------------------
__global__ __launch_bounds__(128) void attn_bf16x3_kernel(
    const __bf16* __restrict__ Qhi, const __bf16* __restrict__ Qlo,
    const __bf16* __restrict__ Khi, const __bf16* __restrict__ Klo,
    const __bf16* __restrict__ Vthi, const __bf16* __restrict__ Vtlo,
    const __bf16* __restrict__ Qbhi, const __bf16* __restrict__ Qblo,
    const __bf16* __restrict__ Kbhi, const __bf16* __restrict__ Kblo,
    const float* __restrict__ mask, float* __restrict__ AO) {
  // staged tiles: [buf][row][40] (80B row stride: 16B aligned, conflict-light)
  __shared__ __bf16 KhiS[2][32][40], KloS[2][32][40];
  __shared__ __bf16 KbhiS[2][32][40], KbloS[2][32][40];
  __shared__ __bf16 VthiS[2][32][40], VtloS[2][32][40];
  __shared__ float Ps[4][16][36];

  const int w = threadIdx.x >> 5;
  const int lane = threadIdx.x & 31;
  const int lr = lane & 15;
  const int hi = lane >> 4;

  const int tile = blockIdx.x * 4 + w;
  const int qt = tile & (Sn / 16 - 1);
  const int bh = tile / (Sn / 16);  // same for all 4 waves (4 | 128)
  const int b = bh / NHn;
  const int h = bh % NHn;
  const int qbase = qt * 16;

  const size_t bhoff = (size_t)bh * Sn * HSn;  // [bh][s][d] planes
  const size_t bhv = (size_t)bh * HSn * Sn;    // [bh][d][s] planes
  const float* maskp = mask + (size_t)b * Sn * Sn;

  // ---- async copy helpers: one b128 per lane ----
  const int crow = lane >> 2;
  const int ccol = (lane & 3) * 16;  // bytes within 64B row
  auto issue_k4 = [&](const __bf16* g, uint32_t lbase) {  // 32x64B contiguous
#pragma unroll
    for (int blk = 0; blk < 4; blk++) {
      uint64_t ga = (uint64_t)(uintptr_t)g + (uint64_t)((blk * 8 + crow) * 64 + ccol);
      uint32_t la = lbase + (uint32_t)((blk * 8 + crow) * 80 + ccol);
      asm volatile("global_load_async_to_lds_b128 %0, %1, off"
                   :: "v"(la), "v"(ga) : "memory");
    }
  };
  auto issue_v2 = [&](const __bf16* g, uint32_t lbase) {  // 16 rows, 4KB stride
#pragma unroll
    for (int blk = 0; blk < 2; blk++) {
      uint64_t ga = (uint64_t)(uintptr_t)g +
                    (uint64_t)(blk * 8 + crow) * (Sn * 2) + ccol;
      uint32_t la = lbase + (uint32_t)((blk * 8 + crow) * 80 + ccol);
      asm volatile("global_load_async_to_lds_b128 %0, %1, off"
                   :: "v"(la), "v"(ga) : "memory");
    }
  };
  auto issue_tiles = [&](int kb, int buf) {  // 6 async b128 per wave
    const size_t ko = bhoff + (size_t)kb * HSn;
    switch (w) {
      case 0:
        issue_k4(Khi + ko, lds_off(&KhiS[buf][0][0]));
        issue_v2(Vthi + bhv + kb, lds_off(&VthiS[buf][0][0]));
        break;
      case 1:
        issue_k4(Klo + ko, lds_off(&KloS[buf][0][0]));
        issue_v2(Vthi + bhv + (size_t)16 * Sn + kb, lds_off(&VthiS[buf][16][0]));
        break;
      case 2:
        issue_k4(Kbhi + ko, lds_off(&KbhiS[buf][0][0]));
        issue_v2(Vtlo + bhv + kb, lds_off(&VtloS[buf][0][0]));
        break;
      default:
        issue_k4(Kblo + ko, lds_off(&KbloS[buf][0][0]));
        issue_v2(Vtlo + bhv + (size_t)16 * Sn + kb, lds_off(&VtloS[buf][16][0]));
        break;
    }
  };

  // B-operand from LDS: elements e<8 -> row[8*hi + e], e>=8 -> row[16 + 8*hi + e-8], 16B-aligned halves
  auto ldsB = [&](const __bf16* p) {
    v8bf a0 = *(const v8bf*)(p - 8 * hi);
    v8bf a1 = *(const v8bf*)(p - 8 * hi + 16);
    v16bf r;
#pragma unroll
    for (int e = 0; e < 8; e++) { r[e] = a0[e]; r[8 + e] = a1[e]; }
    return r;
  };
  // A-operand (16-bit layout): lane = row, d-chunks at 8hi and 16+8hi
  auto loadA = [&](const __bf16* base) {
    const __bf16* p = base + bhoff + (size_t)(qbase + lr) * HSn + 8 * hi;
    v8bf c0 = *(const v8bf*)p;
    v8bf c1 = *(const v8bf*)(p + 16);
    v16bf r;
#pragma unroll
    for (int e = 0; e < 8; e++) { r[e] = c0[e]; r[8 + e] = c1[e]; }
    return r;
  };

  const v16bf qh = loadA(Qhi), ql = loadA(Qlo);
  const v16bf qbh = loadA(Qbhi), qbl = loadA(Qblo);

  float m_run[8], l_run[8];
#pragma unroll
  for (int v = 0; v < 8; v++) { m_run[v] = -3.0e38f; l_run[v] = 0.0f; }
  v8f acc0 = {}, acc1 = {};

  issue_tiles(0, 0);  // prologue

  for (int it = 0; it < Sn / 32; it++) {
    const int kb = it * 32;
    const int buf = it & 1;
    if (it + 1 < Sn / 32) {
      issue_tiles(kb + 32, buf ^ 1);
      asm volatile("s_wait_asynccnt 0x6" ::: "memory");  // drain current buf
      __builtin_prefetch(maskp + (size_t)(qbase + lr) * Sn + kb + 32, 0, 0);
    } else {
      asm volatile("s_wait_asynccnt 0x0" ::: "memory");
    }
    __syncthreads();  // all waves' copies for `buf` visible

    // ---- scores: Q K^T + Qb Kb^T, bf16x3, K=32 per WMMA ----
    v8f sc0 = {}, sc1 = {};
    {
      v16bf kh = ldsB(&KhiS[buf][lr][16 * hi]);
      v16bf kl = ldsB(&KloS[buf][lr][16 * hi]);
      sc0 = wmma_bf(qh, kh, sc0);
      sc0 = wmma_bf(qh, kl, sc0);
      sc0 = wmma_bf(ql, kh, sc0);
      v16bf ch = ldsB(&KbhiS[buf][lr][16 * hi]);
      v16bf cl = ldsB(&KbloS[buf][lr][16 * hi]);
      sc0 = wmma_bf(qbh, ch, sc0);
      sc0 = wmma_bf(qbh, cl, sc0);
      sc0 = wmma_bf(qbl, ch, sc0);
    }
    {
      v16bf kh = ldsB(&KhiS[buf][16 + lr][16 * hi]);
      v16bf kl = ldsB(&KloS[buf][16 + lr][16 * hi]);
      sc1 = wmma_bf(qh, kh, sc1);
      sc1 = wmma_bf(qh, kl, sc1);
      sc1 = wmma_bf(ql, kh, sc1);
      v16bf ch = ldsB(&KbhiS[buf][16 + lr][16 * hi]);
      v16bf cl = ldsB(&KbloS[buf][16 + lr][16 * hi]);
      sc1 = wmma_bf(qbh, ch, sc1);
      sc1 = wmma_bf(qbh, cl, sc1);
      sc1 = wmma_bf(qbl, ch, sc1);
    }

    // ---- online softmax (lane-aligned rows, shuffle reductions) ----
#pragma unroll
    for (int v = 0; v < 8; v++) {
      size_t mrow = (size_t)(qbase + v + 8 * hi) * Sn + kb + lr;
      float s0 = sc0[v] + maskp[mrow];
      float s1 = sc1[v] + maskp[mrow + 16];
      float mt = fmaxf(s0, s1);
#pragma unroll
      for (int off = 1; off < 16; off <<= 1)
        mt = fmaxf(mt, __shfl_xor(mt, off, 32));
      float mnew = fmaxf(m_run[v], mt);
      float scale = __expf(m_run[v] - mnew);
      float p0 = __expf(s0 - mnew);
      float p1 = __expf(s1 - mnew);
      float rt = p0 + p1;
#pragma unroll
      for (int off = 1; off < 16; off <<= 1) rt += __shfl_xor(rt, off, 32);
      l_run[v] = l_run[v] * scale + rt;
      m_run[v] = mnew;
      acc0[v] *= scale;
      acc1[v] *= scale;
      Ps[w][v + 8 * hi][lr] = p0;
      Ps[w][v + 8 * hi][16 + lr] = p1;
    }

    // ---- P: D-layout -> A-layout via LDS, split hi/lo ----
    v16bf ph, pl;
#pragma unroll
    for (int e = 0; e < 8; e++) {
      float f0 = Ps[w][lr][8 * hi + e];
      __bf16 h0 = (__bf16)f0;
      ph[e] = h0;
      pl[e] = (__bf16)(f0 - (float)h0);
      float f1 = Ps[w][lr][16 + 8 * hi + e];
      __bf16 h1 = (__bf16)f1;
      ph[8 + e] = h1;
      pl[8 + e] = (__bf16)(f1 - (float)h1);
    }

    // ---- P @ V from staged transposed V tiles ----
    {
      v16bf vh0 = ldsB(&VthiS[buf][lr][16 * hi]);
      v16bf vl0 = ldsB(&VtloS[buf][lr][16 * hi]);
      acc0 = wmma_bf(ph, vh0, acc0);
      acc0 = wmma_bf(ph, vl0, acc0);
      acc0 = wmma_bf(pl, vh0, acc0);
      v16bf vh1 = ldsB(&VthiS[buf][16 + lr][16 * hi]);
      v16bf vl1 = ldsB(&VtloS[buf][16 + lr][16 * hi]);
      acc1 = wmma_bf(ph, vh1, acc1);
      acc1 = wmma_bf(ph, vl1, acc1);
      acc1 = wmma_bf(pl, vh1, acc1);
    }
    __syncthreads();  // done reading `buf` before anyone refills it
  }

#pragma unroll
  for (int v = 0; v < 8; v++) {
    float inv = 1.0f / l_run[v];
    int s = qbase + v + 8 * hi;
    size_t base = ((size_t)b * Sn + s) * Hn + h * HSn;
    AO[base + lr] = acc0[v] * inv;
    AO[base + 16 + lr] = acc1[v] * inv;
  }
}

// ---------------------------------------------------------------------------
// Kernel 3: out = LayerNorm(AO @ Wf + bf + item)   (exact f32 WMMA)
// ---------------------------------------------------------------------------
__global__ __launch_bounds__(256) void final_ln_kernel(
    const float* __restrict__ AO, const float* __restrict__ item,
    const float* __restrict__ Wf, const float* __restrict__ bf,
    const float* __restrict__ lnw, const float* __restrict__ lnb,
    float* __restrict__ out) {
  __shared__ float Xs[16][132];
  __shared__ float Ys[16][132];

  const int rowbase = blockIdx.x * 16;
  for (int i = threadIdx.x; i < 16 * Hn; i += 256) {
    int r = i >> 7, c = i & 127;
    Xs[r][c] = AO[(size_t)(rowbase + r) * Hn + c];
  }
  __syncthreads();

  const int lane = threadIdx.x & 31;
  const int lr = lane & 15;
  const int hi = lane >> 4;
  const int ncol = (threadIdx.x >> 5) * 16 + lr;

  v8f acc = {};
#pragma unroll
  for (int k = 0; k < Hn; k += 4) {
    v2f a, b;
    a.x = Xs[lr][k + 2 * hi];
    a.y = Xs[lr][k + 2 * hi + 1];
    b.x = Wf[(size_t)(k + 2 * hi) * Hn + ncol];
    b.y = Wf[(size_t)(k + 2 * hi + 1) * Hn + ncol];
    acc = wmma_f32(a, b, acc);
  }

  const float bv_ = bf[ncol];
#pragma unroll
  for (int v = 0; v < 8; v++) {
    int r = rowbase + v + 8 * hi;
    Ys[v + 8 * hi][ncol] = acc[v] + bv_ + item[(size_t)r * Hn + ncol];
  }
  __syncthreads();

  const int r = threadIdx.x >> 4;
  const int c0 = threadIdx.x & 15;
  float vals[8];
  float sum = 0.0f;
#pragma unroll
  for (int j = 0; j < 8; j++) {
    vals[j] = Ys[r][c0 + 16 * j];
    sum += vals[j];
  }
#pragma unroll
  for (int off = 1; off < 16; off <<= 1) sum += __shfl_xor(sum, off, 32);
  float mean = sum * (1.0f / Hn);
  float sq = 0.0f;
#pragma unroll
  for (int j = 0; j < 8; j++) {
    float d = vals[j] - mean;
    sq += d * d;
  }
#pragma unroll
  for (int off = 1; off < 16; off <<= 1) sq += __shfl_xor(sq, off, 32);
  float rstd = rsqrtf(sq * (1.0f / Hn) + EPSL);

  const size_t rowoff = (size_t)(rowbase + r) * Hn;
#pragma unroll
  for (int j = 0; j < 8; j++) {
    int c = c0 + 16 * j;
    out[rowoff + c] = lnw[c] * ((vals[j] - mean) * rstd) + lnb[c];
  }
}

// ---------------------------------------------------------------------------
extern "C" void kernel_launch(void* const* d_in, const int* in_sizes, int n_in,
                              void* d_out, int out_size, void* d_ws,
                              size_t ws_size, hipStream_t stream) {
  const float* item  = (const float*)d_in[0];
  const float* behav = (const float*)d_in[1];
  const float* mask  = (const float*)d_in[2];
  const float* Wq  = (const float*)d_in[3];
  const float* bq  = (const float*)d_in[4];
  const float* Wk  = (const float*)d_in[5];
  const float* bk  = (const float*)d_in[6];
  const float* Wv  = (const float*)d_in[7];
  const float* bv  = (const float*)d_in[8];
  const float* Wqb = (const float*)d_in[9];
  const float* bqb = (const float*)d_in[10];
  const float* Wkb = (const float*)d_in[11];
  const float* bkb = (const float*)d_in[12];
  // d_in[13]/d_in[14] = Wvb/bvb: dead in the reference graph, skipped.
  const float* Wf  = (const float*)d_in[15];
  const float* bf  = (const float*)d_in[16];
  const float* lnw = (const float*)d_in[17];
  const float* lnb = (const float*)d_in[18];
  float* outp = (float*)d_out;

  const size_t N1 = (size_t)Bn * NHn * Sn * HSn;  // 2M elements per plane
  __bf16* Qhi  = (__bf16*)d_ws;
  __bf16* Qlo  = Qhi + N1;
  __bf16* Khi  = Qlo + N1;
  __bf16* Klo  = Khi + N1;
  __bf16* Vthi = Klo + N1;
  __bf16* Vtlo = Vthi + N1;
  __bf16* Qbhi = Vtlo + N1;
  __bf16* Qblo = Qbhi + N1;
  __bf16* Kbhi = Qblo + N1;
  __bf16* Kblo = Kbhi + N1;
  float*  AO   = (float*)(Kblo + N1);  // 40MB bf16 planes + 8MB fp32

  proj_wmma_kernel<<<dim3(Bn * Sn / 16, 5), 256, 0, stream>>>(
      item, behav, Wq, bq, Wk, bk, Wv, bv, Wqb, bqb, Wkb, bkb, Qhi, Qlo, Khi,
      Klo, Vthi, Vtlo, Qbhi, Qblo, Kbhi, Kblo);

  attn_bf16x3_kernel<<<Bn * NHn * (Sn / 16) / 4, 128, 0, stream>>>(
      Qhi, Qlo, Khi, Klo, Vthi, Vtlo, Qbhi, Qblo, Kbhi, Kblo, mask, AO);

  final_ln_kernel<<<Bn * Sn / 16, 256, 0, stream>>>(AO, item, Wf, bf, lnw, lnb,
                                                    outp);
}
